// LocalGeometryEncoding_49435073577117
// MI455X (gfx1250) — matrix operand order repaired and checked by reading, not compiled, not run on hardware
//
#include <hip/hip_runtime.h>

typedef __attribute__((ext_vector_type(16))) _Float16 v16h;
typedef __attribute__((ext_vector_type(8)))  _Float16 v8h;
typedef __attribute__((ext_vector_type(8)))  float    v8f;

#define NVOL    100000
#define KNEI    32
#define NCH     8
#define DIN     256                  // C*K
#define BASE    512
#define NGRID   262144               // 64^3
#define MTILE   32                   // points per block (100000 = 32*3125)
#define XSTRIDE (DIN + 8)            // 264 halves -> 528B rows, 16B aligned
#define HSTRIDE (BASE + 8)           // 520 halves -> 1040B rows, 16B aligned

// ---------------------------------------------------------------------------
// A-matrix fragment (16x32 f16) from a row-major f16 buffer in LDS.
// ISA layout: lane l (0-15) holds row M=l; halves j=0..7 -> K = j (+8 if hi
// lane half), j=8..15 -> K = 16+j-8 (+8 if hi lane half).
// => two aligned 16-byte loads at koff and koff+16.
// ---------------------------------------------------------------------------
__device__ __forceinline__ v16h load_a_frag(const _Float16* __restrict__ buf,
                                            int stride, int m0, int kbase,
                                            int lane) {
  const int m    = m0 + (lane & 15);
  const int koff = kbase + ((lane >> 4) << 3);   // +8 halves for lanes 16..31
  const _Float16* p = buf + m * stride + koff;
  v8h lo = *(const v8h*)(p);
  v8h hi = *(const v8h*)(p + 16);
  return __builtin_shufflevector(lo, hi,
                                 0, 1, 2, 3, 4, 5, 6, 7,
                                 8, 9, 10, 11, 12, 13, 14, 15);
}

// B-matrix (32x16 f16) per-lane K index: j=0..7 -> K = j (+8 if hi lane half),
// j=8..15 -> K = 16+j-8 (+8 if hi lane half) (same as A).
__device__ __forceinline__ int bmap_k(int j, int lane) {
  return j + ((lane >> 4) << 3) + ((j >> 3) << 3);
}

// ---------------------------------------------------------------------------
// Prep: convert W1 (256x512 f32) and W2 (512x32 f32) into f16 workspace laid
// out exactly as per-lane WMMA B fragments, so the main kernel does coalesced
// 32B/lane fragment loads.
//   w1f: frag index = nt*8 + kt   (nt: 0..31 over BASE/16, kt: 0..7 over DIN/32)
//   w2f: frag index = nt*16 + kt  (nt: 0..1  over KNEI/16, kt: 0..15 over BASE/32)
//   element at ((frag*32 + lane)*16 + j)
// ---------------------------------------------------------------------------
__global__ void lge_prep(const float* __restrict__ W1,
                         const float* __restrict__ W2,
                         _Float16* __restrict__ w1f,
                         _Float16* __restrict__ w2f) {
  const int NW1 = (BASE / 16) * (DIN / 32) * 32;   // 8192 threads
  const int NW2 = (KNEI / 16) * (BASE / 32) * 32;  // 1024 threads
  int t = blockIdx.x * blockDim.x + threadIdx.x;
  if (t < NW1) {
    int lane = t & 31;
    int frag = t >> 5;          // nt*8 + kt
    int kt = frag & 7;
    int nt = frag >> 3;
    int n  = nt * 16 + (lane & 15);
#pragma unroll
    for (int j = 0; j < 16; ++j) {
      int k = kt * 32 + bmap_k(j, lane);
      w1f[(size_t)t * 16 + j] = (_Float16)W1[k * BASE + n];
    }
  } else if (t < NW1 + NW2) {
    int u = t - NW1;
    int lane = u & 31;
    int frag = u >> 5;          // nt*16 + kt
    int kt = frag & 15;
    int nt = frag >> 4;
    int n  = nt * 16 + (lane & 15);
#pragma unroll
    for (int j = 0; j < 16; ++j) {
      int k = kt * 32 + bmap_k(j, lane);
      w2f[(size_t)u * 16 + j] = (_Float16)W2[k * KNEI + n];
    }
  }
}

// ---------------------------------------------------------------------------
// Fused: gather -> GEMM1(relu) -> GEMM2, all WMMA f32<-f16, one block = 32 pts
// ---------------------------------------------------------------------------
__global__ __launch_bounds__(256)
void lge_main(const float* __restrict__ enc,     // (8, NGRID)
              const int*   __restrict__ mapping, // (NVOL, 32)
              const float* __restrict__ b1,      // (512)
              const float* __restrict__ b2,      // (32)
              const v16h*  __restrict__ w1f,     // B fragments of W1
              const v16h*  __restrict__ w2f,     // B fragments of W2
              float* __restrict__ out) {         // (NVOL, 32)
  __shared__ _Float16 Xs[MTILE * XSTRIDE];       // ~16.5 KB
  __shared__ _Float16 Hs[MTILE * HSTRIDE];       // ~32.5 KB

  const int tid  = threadIdx.x;
  const int lane = tid & 31;
  const int wid  = tid >> 5;                     // 8 waves
  const int n0   = blockIdx.x * MTILE;

  // ---- Phase 1: masked gather into Xs (f16), X[row][c*32 + k] ----
#pragma unroll
  for (int it = 0; it < (MTILE * KNEI) / 256; ++it) {
    int p   = tid + it * 256;                    // (row, neighbor) pair
    int row = p >> 5;
    int kk  = p & 31;
    int n   = n0 + row;
    int idx = (n < NVOL) ? mapping[n * KNEI + kk] : 0;
    float vals[NCH];
    if (idx != 0) {
#pragma unroll
      for (int c = 0; c < NCH; ++c) vals[c] = enc[c * NGRID + idx];
    } else {
#pragma unroll
      for (int c = 0; c < NCH; ++c) vals[c] = 0.0f;
    }
#pragma unroll
    for (int c = 0; c < NCH; ++c)
      Xs[row * XSTRIDE + c * KNEI + kk] = (_Float16)vals[c];
  }
  __syncthreads();

  // ---- Phase 2: H = relu(X*W1 + b1); 2 M-tiles x 32 N-tiles, 4 N-tiles/wave
#pragma unroll
  for (int ntl = 0; ntl < 4; ++ntl) {
    int nt = wid * 4 + ntl;
    v8f acc0 = {}, acc1 = {};
#pragma unroll
    for (int kt = 0; kt < DIN / 32; ++kt) {
      v16h b  = w1f[(nt * 8 + kt) * 32 + lane];
      v16h a0 = load_a_frag(Xs, XSTRIDE, 0,  kt * 32, lane);
      v16h a1 = load_a_frag(Xs, XSTRIDE, 16, kt * 32, lane);
      acc0 = __builtin_amdgcn_wmma_f32_16x16x32_f16(false, a0, false, b,
                                                    (short)0, acc0, false, false);
      acc1 = __builtin_amdgcn_wmma_f32_16x16x32_f16(false, a1, false, b,
                                                    (short)0, acc1, false, false);
    }
    // C/D layout: lane gives N = lane&15; vgpr r gives M = r + 8*(lane>=16)
    int   nh  = nt * 16 + (lane & 15);
    float bb  = b1[nh];
    int   mhi = (lane >> 4) << 3;
#pragma unroll
    for (int r = 0; r < 8; ++r) {
      float h0 = acc0[r] + bb; h0 = h0 > 0.0f ? h0 : 0.0f;
      float h1 = acc1[r] + bb; h1 = h1 > 0.0f ? h1 : 0.0f;
      Hs[(mhi + r) * HSTRIDE + nh]        = (_Float16)h0;
      Hs[(16 + mhi + r) * HSTRIDE + nh]   = (_Float16)h1;
    }
  }
  __syncthreads();

  // ---- Phase 3: out = H*W2 + b2; 2 M-tiles x 2 N-tiles, one per wave ----
  if (wid < 4) {
    int mt = wid >> 1;
    int nt = wid & 1;
    v8f acc = {};
#pragma unroll
    for (int kt = 0; kt < BASE / 32; ++kt) {
      v16h b = w2f[(nt * 16 + kt) * 32 + lane];
      v16h a = load_a_frag(Hs, HSTRIDE, mt * 16, kt * 32, lane);
      acc = __builtin_amdgcn_wmma_f32_16x16x32_f16(false, a, false, b,
                                                   (short)0, acc, false, false);
    }
    int   nc  = nt * 16 + (lane & 15);
    float bb  = b2[nc];
    int   mhi = (lane >> 4) << 3;
#pragma unroll
    for (int r = 0; r < 8; ++r) {
      int row = n0 + mt * 16 + mhi + r;
      if (row < NVOL) out[row * KNEI + nc] = acc[r] + bb;
    }
  }
}

extern "C" void kernel_launch(void* const* d_in, const int* in_sizes, int n_in,
                              void* d_out, int out_size, void* d_ws, size_t ws_size,
                              hipStream_t stream) {
  const float* enc     = (const float*)d_in[0];   // encoding_g (1,8,64,64,64)
  const int*   mapping = (const int*)  d_in[1];   // (1,NVOL,32)
  const float* W1      = (const float*)d_in[2];   // (256,512)
  const float* b1      = (const float*)d_in[3];   // (512)
  const float* W2      = (const float*)d_in[4];   // (512,32)
  const float* b2      = (const float*)d_in[5];   // (32)
  float* out = (float*)d_out;                     // (1,NVOL,32)

  _Float16* w1f = (_Float16*)d_ws;                // 8192*16 halves = 256 KB
  _Float16* w2f = w1f + (size_t)8192 * 16;        // 1024*16 halves = 32 KB

  lge_prep<<<(8192 + 1024 + 255) / 256, 256, 0, stream>>>(W1, W2, w1f, w2f);
  lge_main<<<NVOL / MTILE, 256, 0, stream>>>(enc, mapping, b1, b2,
                                             (const v16h*)w1f,
                                             (const v16h*)w2f, out);
}
